// Model_57269093925194
// MI455X (gfx1250) — compile-verified, matrix operand order checked by reading, not run on hardware
//
#include <hip/hip_runtime.h>
#include <hip/hip_bf16.h>
#include <math.h>

typedef _Float16 half_t;
typedef __attribute__((ext_vector_type(16))) _Float16 v16h;
typedef __attribute__((ext_vector_type(8)))  _Float16 v8h;
typedef __attribute__((ext_vector_type(4)))  _Float16 v4h;
typedef __attribute__((ext_vector_type(8)))  float    v8f;

#define N_NODES 50000
#define N_EDGES 800000
#define FDIM 53
#define NB 128
#define NH 4
#define GATF (NH*FDIM)   /* 212 */
#define F2   (2*FDIM)    /* 106 */
#define E2   (N_EDGES + N_NODES)   /* edges + self loops */

#define CDIV(a,b) (((a)+(b)-1)/(b))

#define CAT16(lo, hi) __builtin_shufflevector((lo), (hi), 0,1,2,3,4,5,6,7,8,9,10,11,12,13,14,15)

// ---------- order-preserving float<->uint encoding (for atomic max) ----------
__device__ __forceinline__ unsigned enc_f32(float f) {
  unsigned u = __float_as_uint(f);
  return (u & 0x80000000u) ? ~u : (u | 0x80000000u);
}
__device__ __forceinline__ float dec_f32(unsigned u) {
  unsigned b = (u & 0x80000000u) ? (u ^ 0x80000000u) : ~u;
  return __uint_as_float(b);
}

// ---------- f32 -> f16 weight conversion ----------
__global__ void k_f32_to_f16(const float* __restrict__ s, half_t* __restrict__ d, int n) {
  int i = blockIdx.x * blockDim.x + threadIdx.x;
  if (i < n) d[i] = (half_t)s[i];
}

// ---------- generic WMMA GEMM: C[M,N] = act(A[M,K] @ B[K,N] + bias) ----------
// A: f32 (row stride lda), B: f16 (row stride N), C: f32 (row stride ldc).
// grid = (ceil(M/64), ceil(N/64)), block = 128 (4 waves); each wave owns a
// 16-row strip and 4 16x16 col tiles.
// LDS: double-buffered A tile [64][32] and B tile TRANSPOSED [64 cols][32 k],
// rows padded to 40 halfs (80 B) -> 16-B aligned b128 fragment loads,
// conflict-free banks. Interior tiles use unguarded staging (batched loads);
// edge tiles take the guarded path.
#define GS 40
__global__ void __launch_bounds__(128)
k_gemm_wmma(const float* __restrict__ A, int lda,
            const half_t* __restrict__ Bw,
            const float* __restrict__ bias,
            float* __restrict__ C, int ldc,
            int M, int K, int N, int relu)
{
  __shared__ __align__(16) half_t As[2][64][GS];
  __shared__ __align__(16) half_t Bs[2][64][GS];   // [col][k]
  const int tid  = threadIdx.x;
  const int wave = tid >> 5;
  const int lane = tid & 31;
  const int rowBase = blockIdx.x * 64;
  const int colBase = blockIdx.y * 64;

  v8f acc0 = {}, acc1 = {}, acc2 = {}, acc3 = {};
  v8f* acc[4] = {&acc0, &acc1, &acc2, &acc3};

  const int rl  = (wave << 4) | (lane & 15);
  const int klo = (lane < 16) ? 0 : 8;

  auto loadA = [&](int k0, int buf) {
    if ((rowBase + 64 <= M) && (k0 + 32 <= K)) {
      // unguarded: 4 consecutive floats per thread -> one 8-B LDS store
      for (int i = tid; i < 512; i += 128) {
        int r = i >> 3, kq = (i & 7) << 2;
        const float* ap = &A[(long long)(rowBase + r) * lda + (k0 + kq)];
        v4h h;
        h[0] = (half_t)ap[0]; h[1] = (half_t)ap[1];
        h[2] = (half_t)ap[2]; h[3] = (half_t)ap[3];
        *(v4h*)&As[buf][r][kq] = h;
      }
    } else {
      for (int i = tid; i < 64 * 32; i += 128) {
        int r = i >> 5, kk = i & 31;
        int gr = rowBase + r, gk = k0 + kk;
        float v = (gr < M && gk < K) ? A[(long long)gr * lda + gk] : 0.f;
        As[buf][r][kk] = (half_t)v;
      }
    }
  };
  auto loadB = [&](int k0, int buf) {
    if ((colBase + 64 <= N) && (k0 + 32 <= K)) {
      for (int i = tid; i < 32 * 64; i += 128) {
        int c = i & 63, kk = i >> 6;
        Bs[buf][c][kk] = Bw[(k0 + kk) * N + (colBase + c)];
      }
    } else {
      for (int i = tid; i < 32 * 64; i += 128) {
        int c = i & 63, kk = i >> 6;
        int gk = k0 + kk, gc = colBase + c;
        half_t v = (half_t)0.f;
        if (gk < K && gc < N) v = Bw[gk * N + gc];
        Bs[buf][c][kk] = v;
      }
    }
  };

  loadA(0, 0);
  loadB(0, 0);
  __syncthreads();

  int p = 0;
  for (int k0 = 0; k0 < K; k0 += 32) {
    v8h alo = *(const v8h*)&As[p][rl][klo];
    v8h ahi = *(const v8h*)&As[p][rl][klo + 16];
    v16h a = CAT16(alo, ahi);
#pragma unroll
    for (int j = 0; j < 4; ++j) {
      int cl = (j << 4) | (lane & 15);
      v8h blo = *(const v8h*)&Bs[p][cl][klo];
      v8h bhi = *(const v8h*)&Bs[p][cl][klo + 16];
      v16h b = CAT16(blo, bhi);
      *acc[j] = __builtin_amdgcn_wmma_f32_16x16x32_f16(
          false, a, false, b, (short)0, *acc[j], false, false);
    }
    if (k0 + 32 < K) {           // stage next tile while this one computes
      loadA(k0 + 32, p ^ 1);
      loadB(k0 + 32, p ^ 1);
    }
    __syncthreads();
    p ^= 1;
  }

  const int rbase = rowBase + (wave << 4) + ((lane < 16) ? 0 : 8);
  for (int j = 0; j < 4; ++j) {
    int col = colBase + (j << 4) + (lane & 15);
    if (col >= N) continue;
    float bb = bias ? bias[col] : 0.f;
    for (int r = 0; r < 8; ++r) {
      int row = rbase + r;
      if (row < M) {
        float v = (*acc[j])[r] + bb;
        if (relu) v = fmaxf(v, 0.f);
        C[(long long)row * ldc + col] = v;
      }
    }
  }
}

// ---------- fused EdgeConv: per-edge 2-layer MLP + encoded atomic segment max ----------
// block = 128 threads (4 waves), 64 edges per block.
// feat/h1: [64 rows][128 k + 8 pad] (272-B rows: aligned + conflict-free).
// wpan: weight panel TRANSPOSED [112 cols][128 k] with 16-B chunk XOR swizzle
// (chunk ^= col&15) for conflict-free b128 column-fragment loads without pad.
// Total LDS = 17408*2 + 28672 + 512 = 64000 B.
#define FK 136
__global__ void __launch_bounds__(128)
k_edgeconv(const float* __restrict__ x,
           const int* __restrict__ src, const int* __restrict__ dst,
           const half_t* __restrict__ w1, const float* __restrict__ b1,
           const half_t* __restrict__ w2, const float* __restrict__ b2,
           unsigned* __restrict__ out_enc)
{
  __shared__ __align__(16) half_t feat[64][FK];
  __shared__ __align__(16) half_t h1[64][FK];
  __shared__ __align__(16) half_t wpan[112][128];
  __shared__ int sloc[64], dloc[64];

  const int tid  = threadIdx.x;
  const int wave = tid >> 5;
  const int lane = tid & 31;
  const int e0 = blockIdx.x * 64;

  if (tid < 64) {
    int e = e0 + tid;
    int s = 0, d = 0;
    if (e < N_EDGES) { s = src[e]; d = dst[e]; }
    sloc[tid] = s; dloc[tid] = d;
  }
  // w1 -> wpan (transposed + chunk-swizzled); lanes sweep col fastest
  for (int i = tid; i < 112 * 128; i += 128) {
    int c = i % 112, k = i / 112;
    half_t v = (k < F2 && c < F2) ? w1[k * F2 + c] : (half_t)0.f;
    int kk = ((((k >> 3) ^ (c & 15)) & 15) << 3) | (k & 7);
    wpan[c][kk] = v;
  }
  __syncthreads();

  // gather feat = [x_i, x_j - x_i] (i = dst, j = src), zero-pad; zero h1
  for (int i = tid; i < 64 * 128; i += 128) {
    int r = i >> 7, k = i & 127;
    int e = e0 + r;
    float v = 0.f;
    if (e < N_EDGES) {
      int s = sloc[r], d = dloc[r];
      if (k < FDIM)      v = x[d * FDIM + k];
      else if (k < F2) { int f = k - FDIM; v = x[s * FDIM + f] - x[d * FDIM + f]; }
    }
    feat[r][k] = (half_t)v;
    h1[r][k]   = (half_t)0.f;
  }
  __syncthreads();

  const int rl   = (wave << 4) | (lane & 15);
  const int klo  = (lane < 16) ? 0 : 8;
  const int rowb = (wave << 4) + ((lane < 16) ? 0 : 8);
  const int cswz = (lane & 15);

  // layer 1: h1 = relu(feat @ w1 + b1)
  for (int n0 = 0; n0 < 112; n0 += 16) {
    int cl = n0 + (lane & 15);
    v8f acc = {};
#pragma unroll
    for (int k0 = 0; k0 < 128; k0 += 32) {
      v8h alo = *(const v8h*)&feat[rl][k0 + klo];
      v8h ahi = *(const v8h*)&feat[rl][k0 + klo + 16];
      int kb0 = (((k0 + klo) >> 3) ^ cswz) << 3;
      int kb1 = (((k0 + klo + 16) >> 3) ^ cswz) << 3;
      v8h blo = *(const v8h*)&wpan[cl][kb0];
      v8h bhi = *(const v8h*)&wpan[cl][kb1];
      v16h a = CAT16(alo, ahi);
      v16h b = CAT16(blo, bhi);
      acc = __builtin_amdgcn_wmma_f32_16x16x32_f16(false, a, false, b, (short)0, acc, false, false);
    }
    float bb = (cl < F2) ? b1[cl] : 0.f;
    for (int r = 0; r < 8; ++r)
      h1[rowb + r][cl] = (half_t)fmaxf(acc[r] + bb, 0.f);
  }
  __syncthreads();

  // w2 -> wpan
  for (int i = tid; i < 112 * 128; i += 128) {
    int c = i % 112, k = i / 112;
    half_t v = (k < F2 && c < F2) ? w2[k * F2 + c] : (half_t)0.f;
    int kk = ((((k >> 3) ^ (c & 15)) & 15) << 3) | (k & 7);
    wpan[c][kk] = v;
  }
  __syncthreads();

  // layer 2: out = h1 @ w2 + b2 ; encoded atomic max into dst rows
  for (int n0 = 0; n0 < 112; n0 += 16) {
    int cl = n0 + (lane & 15);
    v8f acc = {};
#pragma unroll
    for (int k0 = 0; k0 < 128; k0 += 32) {
      v8h alo = *(const v8h*)&h1[rl][k0 + klo];
      v8h ahi = *(const v8h*)&h1[rl][k0 + klo + 16];
      int kb0 = (((k0 + klo) >> 3) ^ cswz) << 3;
      int kb1 = (((k0 + klo + 16) >> 3) ^ cswz) << 3;
      v8h blo = *(const v8h*)&wpan[cl][kb0];
      v8h bhi = *(const v8h*)&wpan[cl][kb1];
      v16h a = CAT16(alo, ahi);
      v16h b = CAT16(blo, bhi);
      acc = __builtin_amdgcn_wmma_f32_16x16x32_f16(false, a, false, b, (short)0, acc, false, false);
    }
    if (cl < F2) {
      float bb = b2[cl];
      for (int r = 0; r < 8; ++r) {
        int e = e0 + rowb + r;
        if (e < N_EDGES) {
          float v = acc[r] + bb;
          atomicMax(&out_enc[(long long)dloc[rowb + r] * F2 + cl], enc_f32(v));
        }
      }
    }
  }
}

// decode encoded max, fold where(isfinite,·,0) and outer ReLU; in-place u32 -> f32
__global__ void k_ec_decode(unsigned* __restrict__ buf) {
  int i = blockIdx.x * blockDim.x + threadIdx.x;
  if (i >= N_NODES * F2) return;
  unsigned u = buf[i];
  float v = (u == 0u) ? 0.f : dec_f32(u);
  if (!(v > 0.f)) v = 0.f;   // relu; also kills NaN / -inf / untouched
  ((float*)buf)[i] = v;
}

// ---------- GAT edge phase ----------
__global__ void k_gat_al(const float* __restrict__ xh, const float* __restrict__ asrc,
                         const float* __restrict__ adst,
                         float* __restrict__ als, float* __restrict__ ald) {
  int i = blockIdx.x * blockDim.x + threadIdx.x;
  if (i >= N_NODES * NH) return;
  int n = i >> 2, h = i & 3;
  const float* row = xh + (long long)n * GATF + h * FDIM;
  float s = 0.f, d = 0.f;
  for (int f = 0; f < FDIM; ++f) {
    s += row[f] * asrc[h * FDIM + f];
    d += row[f] * adst[h * FDIM + f];
  }
  als[i] = s; ald[i] = d;
}

__global__ void k_gat_edge_max(const int* __restrict__ src, const int* __restrict__ dst,
                               const float* __restrict__ als, const float* __restrict__ ald,
                               unsigned* __restrict__ menc) {
  int e = blockIdx.x * blockDim.x + threadIdx.x;
  if (e >= E2) return;
  int s, d;
  if (e < N_EDGES) { s = src[e]; d = dst[e]; } else { s = d = e - N_EDGES; }
  for (int h = 0; h < NH; ++h) {
    float v = als[s * NH + h] + ald[d * NH + h];
    v = (v > 0.f) ? v : 0.2f * v;
    atomicMax(&menc[d * NH + h], enc_f32(v));
  }
}

__global__ void k_gat_edge_ez(const int* __restrict__ src, const int* __restrict__ dst,
                              const float* __restrict__ als, const float* __restrict__ ald,
                              const unsigned* __restrict__ menc,
                              float* __restrict__ exbuf, float* __restrict__ zbuf) {
  int e = blockIdx.x * blockDim.x + threadIdx.x;
  if (e >= E2) return;
  int s, d;
  if (e < N_EDGES) { s = src[e]; d = dst[e]; } else { s = d = e - N_EDGES; }
  for (int h = 0; h < NH; ++h) {
    float v = als[s * NH + h] + ald[d * NH + h];
    v = (v > 0.f) ? v : 0.2f * v;
    float ex = expf(v - dec_f32(menc[d * NH + h]));
    exbuf[e * NH + h] = ex;
    atomicAdd(&zbuf[d * NH + h], ex);
  }
}

__global__ void k_gat_edge_msg(const int* __restrict__ src, const int* __restrict__ dst,
                               const float* __restrict__ exbuf, const float* __restrict__ xh,
                               float* __restrict__ outg) {
  int i = blockIdx.x * blockDim.x + threadIdx.x;
  if (i >= E2 * FDIM) return;
  int e = i / FDIM, f = i % FDIM;
  int s, d;
  if (e < N_EDGES) { s = src[e]; d = dst[e]; } else { s = d = e - N_EDGES; }
  for (int h = 0; h < NH; ++h)
    atomicAdd(&outg[(long long)d * GATF + h * FDIM + f],
              exbuf[e * NH + h] * xh[(long long)s * GATF + h * FDIM + f]);
}

__global__ void k_gat_final(float* __restrict__ outg, const float* __restrict__ zbuf,
                            const float* __restrict__ bias) {
  int i = blockIdx.x * blockDim.x + threadIdx.x;
  if (i >= N_NODES * GATF) return;
  int n = i / GATF, f = i % GATF, h = f / FDIM;
  float v = outg[i] / (zbuf[n * NH + h] + 1e-16f) + bias[f];
  outg[i] = fmaxf(v, 0.f);
}

// ---------- GIN scatter ----------
__global__ void k_gin_scatter(const int* __restrict__ src, const int* __restrict__ dst,
                              const float* __restrict__ x, float* __restrict__ agg) {
  int i = blockIdx.x * blockDim.x + threadIdx.x;
  if (i >= N_EDGES * FDIM) return;
  int e = i / FDIM, f = i % FDIM;
  atomicAdd(&agg[(long long)dst[e] * FDIM + f], x[(long long)src[e] * FDIM + f]);
}
__global__ void k_gin_self(const float* __restrict__ x, float* __restrict__ agg) {
  int i = blockIdx.x * blockDim.x + threadIdx.x;
  if (i < N_NODES * FDIM) agg[i] += x[i];
}

// ---------- pooling ----------
__global__ void k_count(const int* __restrict__ batch, float* __restrict__ cnt) {
  int i = blockIdx.x * blockDim.x + threadIdx.x;
  if (i < N_NODES) atomicAdd(&cnt[batch[i]], 1.f);
}
__global__ void k_pool_accum(const float* __restrict__ feat, int fdim,
                             const int* __restrict__ batch, float* __restrict__ pool) {
  int i = blockIdx.x * blockDim.x + threadIdx.x;
  if (i >= N_NODES * fdim) return;
  int n = i / fdim, f = i % fdim;
  atomicAdd(&pool[batch[n] * fdim + f], feat[i]);
}
__global__ void k_pool_div(float* __restrict__ pool, const float* __restrict__ cnt, int fdim) {
  int i = blockIdx.x * blockDim.x + threadIdx.x;
  if (i >= NB * fdim) return;
  pool[i] /= fmaxf(cnt[i / fdim], 1.f);
}

// ---------- final 64 -> 1 + sigmoid ----------
__global__ void k_final(const float* __restrict__ h, const float* __restrict__ w,
                        const float* __restrict__ b, float* __restrict__ out) {
  int i = blockIdx.x * blockDim.x + threadIdx.x;
  if (i >= NB) return;
  float s = 0.f;
  for (int k = 0; k < 64; ++k) s += h[i * 64 + k] * w[k];
  s += b[0];
  out[i] = 1.f / (1.f + expf(-s));
}

// ---------- host-side helpers ----------
static void gemm(hipStream_t st, const float* A, int lda, const half_t* B,
                 const float* bias, float* C, int ldc, int M, int K, int N, int relu) {
  dim3 g(CDIV(M, 64), CDIV(N, 64));
  k_gemm_wmma<<<g, 128, 0, st>>>(A, lda, B, bias, C, ldc, M, K, N, relu);
}

extern "C" void kernel_launch(void* const* d_in, const int* in_sizes, int n_in,
                              void* d_out, int out_size, void* d_ws, size_t ws_size,
                              hipStream_t stream) {
  (void)in_sizes; (void)n_in; (void)out_size; (void)ws_size;
  const float* x        = (const float*)d_in[0];
  const int*   eidx     = (const int*)d_in[1];
  const int*   batch    = (const int*)d_in[2];
  const float* gat_w    = (const float*)d_in[3];
  const float* gat_asrc = (const float*)d_in[4];
  const float* gat_adst = (const float*)d_in[5];
  const float* gat_bias = (const float*)d_in[6];
  const float* gin_w1 = (const float*)d_in[7];  const float* gin_b1 = (const float*)d_in[8];
  const float* gin_w2 = (const float*)d_in[9];  const float* gin_b2 = (const float*)d_in[10];
  const float* ec_w1  = (const float*)d_in[11]; const float* ec_b1  = (const float*)d_in[12];
  const float* ec_w2  = (const float*)d_in[13]; const float* ec_b2  = (const float*)d_in[14];
  const float* fg1_w = (const float*)d_in[15]; const float* fg1_b = (const float*)d_in[16];
  const float* fg2_w = (const float*)d_in[17]; const float* fg2_b = (const float*)d_in[18];
  const float* fg3_w = (const float*)d_in[19]; const float* fg3_b = (const float*)d_in[20];
  const float* fg4_w = (const float*)d_in[21]; const float* fg4_b = (const float*)d_in[22];
  const float* fg5_w = (const float*)d_in[23]; const float* fg5_b = (const float*)d_in[24];
  const float* fg6_w = (const float*)d_in[25]; const float* fg6_b = (const float*)d_in[26];
  const float* fc1_w = (const float*)d_in[27]; const float* fc1_b = (const float*)d_in[28];
  const float* fc2_w = (const float*)d_in[29]; const float* fc2_b = (const float*)d_in[30];
  const float* out_w = (const float*)d_in[31]; const float* out_b = (const float*)d_in[32];

  const int* src = eidx;
  const int* dst = eidx + N_EDGES;

  // ---- carve workspace (deterministic every call) ----
  char* p = (char*)d_ws;
  auto take = [&](size_t bytes) { void* r = (void*)p; p += (bytes + 255) & ~(size_t)255; return r; };

  half_t* wh_gat  = (half_t*)take(sizeof(half_t) * FDIM * GATF);
  half_t* wh_gin1 = (half_t*)take(sizeof(half_t) * FDIM * F2);
  half_t* wh_gin2 = (half_t*)take(sizeof(half_t) * F2 * F2);
  half_t* wh_ec1  = (half_t*)take(sizeof(half_t) * F2 * F2);
  half_t* wh_ec2  = (half_t*)take(sizeof(half_t) * F2 * F2);
  half_t* wh_fg1  = (half_t*)take(sizeof(half_t) * GATF * 256);
  half_t* wh_fg2  = (half_t*)take(sizeof(half_t) * 256 * 128);
  half_t* wh_fg3  = (half_t*)take(sizeof(half_t) * F2 * 256);
  half_t* wh_fg4  = (half_t*)take(sizeof(half_t) * 256 * 128);
  half_t* wh_fg5  = (half_t*)take(sizeof(half_t) * F2 * 256);
  half_t* wh_fg6  = (half_t*)take(sizeof(half_t) * 256 * 128);
  half_t* wh_fc1  = (half_t*)take(sizeof(half_t) * 384 * 128);
  half_t* wh_fc2  = (half_t*)take(sizeof(half_t) * 128 * 64);

  float*    xh    = (float*)take(sizeof(float) * (size_t)N_NODES * GATF);
  float*    als   = (float*)take(sizeof(float) * N_NODES * NH);
  float*    ald   = (float*)take(sizeof(float) * N_NODES * NH);
  unsigned* menc  = (unsigned*)take(sizeof(unsigned) * N_NODES * NH);
  float*    zbuf  = (float*)take(sizeof(float) * N_NODES * NH);
  float*    exbuf = (float*)take(sizeof(float) * (size_t)E2 * NH);
  float*    outg  = (float*)take(sizeof(float) * (size_t)N_NODES * GATF);
  float*    agg   = (float*)take(sizeof(float) * (size_t)N_NODES * FDIM);
  float*    ginh1 = (float*)take(sizeof(float) * (size_t)N_NODES * F2);
  float*    ginout= (float*)take(sizeof(float) * (size_t)N_NODES * F2);
  unsigned* ecenc = (unsigned*)take(sizeof(unsigned) * (size_t)N_NODES * F2);
  float*    pg    = (float*)take(sizeof(float) * NB * GATF);
  float*    pi    = (float*)take(sizeof(float) * NB * F2);
  float*    pe    = (float*)take(sizeof(float) * NB * F2);
  float*    cnt   = (float*)take(sizeof(float) * NB);
  float*    t1    = (float*)take(sizeof(float) * NB * 256);
  float*    t2    = (float*)take(sizeof(float) * NB * 256);
  float*    t3    = (float*)take(sizeof(float) * NB * 256);
  float*    cc    = (float*)take(sizeof(float) * NB * 384);
  float*    hf1   = (float*)take(sizeof(float) * NB * 128);
  float*    hf2   = (float*)take(sizeof(float) * NB * 64);

  // ---- weight conversion to f16 ----
  struct { const float* s; half_t* d; int n; } cv[] = {
    {gat_w, wh_gat, FDIM*GATF}, {gin_w1, wh_gin1, FDIM*F2}, {gin_w2, wh_gin2, F2*F2},
    {ec_w1, wh_ec1, F2*F2}, {ec_w2, wh_ec2, F2*F2},
    {fg1_w, wh_fg1, GATF*256}, {fg2_w, wh_fg2, 256*128},
    {fg3_w, wh_fg3, F2*256},  {fg4_w, wh_fg4, 256*128},
    {fg5_w, wh_fg5, F2*256},  {fg6_w, wh_fg6, 256*128},
    {fc1_w, wh_fc1, 384*128}, {fc2_w, wh_fc2, 128*64},
  };
  for (auto& c : cv)
    k_f32_to_f16<<<CDIV(c.n, 256), 256, 0, stream>>>(c.s, c.d, c.n);

  // ---- zero accumulators (graph-capturable async memsets) ----
  hipMemsetAsync(menc, 0, sizeof(unsigned) * N_NODES * NH, stream);
  hipMemsetAsync(zbuf, 0, sizeof(float) * N_NODES * NH, stream);
  hipMemsetAsync(outg, 0, sizeof(float) * (size_t)N_NODES * GATF, stream);
  hipMemsetAsync(agg,  0, sizeof(float) * (size_t)N_NODES * FDIM, stream);
  hipMemsetAsync(ecenc,0, sizeof(unsigned) * (size_t)N_NODES * F2, stream);
  hipMemsetAsync(pg,   0, sizeof(float) * NB * GATF, stream);
  hipMemsetAsync(pi,   0, sizeof(float) * NB * F2, stream);
  hipMemsetAsync(pe,   0, sizeof(float) * NB * F2, stream);
  hipMemsetAsync(cnt,  0, sizeof(float) * NB, stream);

  // ---- GAT branch ----
  gemm(stream, x, FDIM, wh_gat, nullptr, xh, GATF, N_NODES, FDIM, GATF, 0);
  k_gat_al<<<CDIV(N_NODES * NH, 256), 256, 0, stream>>>(xh, gat_asrc, gat_adst, als, ald);
  k_gat_edge_max<<<CDIV(E2, 256), 256, 0, stream>>>(src, dst, als, ald, menc);
  k_gat_edge_ez<<<CDIV(E2, 256), 256, 0, stream>>>(src, dst, als, ald, menc, exbuf, zbuf);
  k_gat_edge_msg<<<CDIV(E2 * FDIM, 256), 256, 0, stream>>>(src, dst, exbuf, xh, outg);
  k_gat_final<<<CDIV(N_NODES * GATF, 256), 256, 0, stream>>>(outg, zbuf, gat_bias);

  // ---- GIN branch ----
  k_gin_scatter<<<CDIV(N_EDGES * FDIM, 256), 256, 0, stream>>>(src, dst, x, agg);
  k_gin_self<<<CDIV(N_NODES * FDIM, 256), 256, 0, stream>>>(x, agg);
  gemm(stream, agg, FDIM, wh_gin1, gin_b1, ginh1, F2, N_NODES, FDIM, F2, 1);
  gemm(stream, ginh1, F2, wh_gin2, gin_b2, ginout, F2, N_NODES, F2, F2, 1);

  // ---- EdgeConv branch (fused WMMA MLP + encoded atomic max) ----
  k_edgeconv<<<CDIV(N_EDGES, 64), 128, 0, stream>>>(x, src, dst, wh_ec1, ec_b1, wh_ec2, ec_b2, ecenc);
  k_ec_decode<<<CDIV(N_NODES * F2, 256), 256, 0, stream>>>(ecenc);

  // ---- global mean pool ----
  k_count<<<CDIV(N_NODES, 256), 256, 0, stream>>>(batch, cnt);
  k_pool_accum<<<CDIV(N_NODES * GATF, 256), 256, 0, stream>>>(outg, GATF, batch, pg);
  k_pool_accum<<<CDIV(N_NODES * F2, 256), 256, 0, stream>>>(ginout, F2, batch, pi);
  k_pool_accum<<<CDIV(N_NODES * F2, 256), 256, 0, stream>>>((const float*)ecenc, F2, batch, pe);
  k_pool_div<<<CDIV(NB * GATF, 256), 256, 0, stream>>>(pg, cnt, GATF);
  k_pool_div<<<CDIV(NB * F2, 256), 256, 0, stream>>>(pi, cnt, F2);
  k_pool_div<<<CDIV(NB * F2, 256), 256, 0, stream>>>(pe, cnt, F2);

  // ---- per-branch heads (concat built via ldc = 384 with column offsets) ----
  gemm(stream, pg, GATF, wh_fg1, fg1_b, t1, 256, NB, GATF, 256, 1);
  gemm(stream, t1, 256, wh_fg2, fg2_b, cc + 0,   384, NB, 256, 128, 1);
  gemm(stream, pi, F2,  wh_fg3, fg3_b, t2, 256, NB, F2, 256, 1);
  gemm(stream, t2, 256, wh_fg4, fg4_b, cc + 128, 384, NB, 256, 128, 1);
  gemm(stream, pe, F2,  wh_fg5, fg5_b, t3, 256, NB, F2, 256, 1);
  gemm(stream, t3, 256, wh_fg6, fg6_b, cc + 256, 384, NB, 256, 128, 1);

  // ---- final MLP ----
  gemm(stream, cc, 384, wh_fc1, fc1_b, hf1, 128, NB, 384, 128, 1);
  gemm(stream, hf1, 128, wh_fc2, fc2_b, hf2, 64, NB, 128, 64, 1);
  k_final<<<CDIV(NB, 64), 64, 0, stream>>>(hf2, out_w, out_b, (float*)d_out);
}
